// GAT_21045339750566
// MI455X (gfx1250) — matrix operand order repaired and checked by reading, not compiled, not run on hardware
//
#include <hip/hip_runtime.h>
#include <hip/hip_bf16.h>

#define N_NODES 50000
#define N_EDGES 800000
#define F_IN 256
#define HID 64
#define HEADS 4
#define HD 256        // HEADS*HID == F_IN
#define NCLASS 40
#define NCLASS_PAD 64

typedef __attribute__((ext_vector_type(8)))  float  v8f;
typedef __attribute__((ext_vector_type(8)))  __bf16 v8bf;
typedef __attribute__((ext_vector_type(16))) __bf16 v16bf;

union ABu { struct { v8bf lo, hi; } p; v16bf v; };

// ---------------- f32 -> bf16 cast ----------------
__global__ void gat_f32_to_bf16(const float* __restrict__ src,
                                __bf16* __restrict__ dst, int total) {
  int i = blockIdx.x * blockDim.x + threadIdx.x;
  if (i >= total) return;
  dst[i] = (__bf16)src[i];
}

// ---------------- weight pack into WMMA B-fragment layout ----------------
// Packed layout: [kt][nt][lane(32)][16 elems], where for lane L:
//   n = nt*16 + (L&15),  k = kt*32 + (L>>4)*16 + i   (i = element 0..15)
// i.e. contiguous k order, the same order gat_gemm_bf16 loads its A fragment in.
__global__ void gat_pack_w(const float* __restrict__ W, __bf16* __restrict__ Wp,
                           int Cin, int Cp, int K) {
  int idx = blockIdx.x * blockDim.x + threadIdx.x;
  if (idx >= K * Cp) return;
  int i    = idx & 15;
  int lane = (idx >> 4) & 31;
  int rest = idx >> 9;            // 32 lanes * 16 elems = 512
  int NT   = Cp >> 4;
  int nt   = rest % NT;
  int kt   = rest / NT;
  int k = kt * 32 + ((lane >> 4) << 4) + i;
  int n = nt * 16 + (lane & 15);
  float val = (n < Cin) ? W[(size_t)k * Cin + n] : 0.0f;
  Wp[idx] = (__bf16)val;
}

// ---------------- bf16 WMMA GEMM: C[M,Ncols] = A[M,256] @ Bpacked ----------------
// One wave computes a 16x64 tile (4 accumulators), reusing the A fragment.
__global__ __launch_bounds__(256) void gat_gemm_bf16(
    const __bf16* __restrict__ A, const __bf16* __restrict__ Bp,
    float* __restrict__ C, int M, int Ncols) {
  const int K = 256;
  int wave = blockIdx.x * 8 + (threadIdx.x >> 5);
  int lane = threadIdx.x & 31;
  int NG = Ncols >> 6;                 // 64-col groups
  int tiles = (M >> 4) * NG;
  if (wave >= tiles) return;
  int mt = wave / NG;
  int ng = wave - mt * NG;
  int khi = lane >> 4;
  const __bf16* arow = A + (size_t)(mt * 16 + (lane & 15)) * K;
  int NT = Ncols >> 4;

  v8f acc0 = {}, acc1 = {}, acc2 = {}, acc3 = {};
#pragma unroll
  for (int kk = 0; kk < K / 32; ++kk) {
    // A fragment: lane L holds row (L&15); K = khi*16..khi*16+15
    ABu au;
    au.p.lo = *(const v8bf*)(arow + kk * 32 + khi * 16);
    au.p.hi = *(const v8bf*)(arow + kk * 32 + khi * 16 + 8);
    const __bf16* bb = Bp + (((size_t)(kk * NT + ng * 4) * 32) + lane) * 16;
    v16bf b0 = *(const v16bf*)(bb);
    v16bf b1 = *(const v16bf*)(bb + 512);
    v16bf b2 = *(const v16bf*)(bb + 1024);
    v16bf b3 = *(const v16bf*)(bb + 1536);
    acc0 = __builtin_amdgcn_wmma_f32_16x16x32_bf16(false, au.v, false, b0, (short)0, acc0, false, false);
    acc1 = __builtin_amdgcn_wmma_f32_16x16x32_bf16(false, au.v, false, b1, (short)0, acc1, false, false);
    acc2 = __builtin_amdgcn_wmma_f32_16x16x32_bf16(false, au.v, false, b2, (short)0, acc2, false, false);
    acc3 = __builtin_amdgcn_wmma_f32_16x16x32_bf16(false, au.v, false, b3, (short)0, acc3, false, false);
  }
  // C/D layout: VGPR j -> row mt*16 + j + khi*8, col = base + (lane&15)
  float* crow = C + (size_t)(mt * 16 + khi * 8) * Ncols + ng * 64 + (lane & 15);
#pragma unroll
  for (int j = 0; j < 8; ++j) {
    crow[(size_t)j * Ncols]      = acc0[j];
    crow[(size_t)j * Ncols + 16] = acc1[j];
    crow[(size_t)j * Ncols + 32] = acc2[j];
    crow[(size_t)j * Ncols + 48] = acc3[j];
  }
}

// ---------------- attention logits el/er per (node, head) ----------------
__global__ void gat_eler(const float* __restrict__ h, const float* __restrict__ al,
                         const float* __restrict__ ar, float* __restrict__ el,
                         float* __restrict__ er, int N, int H, int D, int stride) {
  int i = blockIdx.x * blockDim.x + threadIdx.x;
  if (i >= N * H) return;
  int n = i / H, hh = i - n * H;
  const float* hp = h + (size_t)n * stride + hh * D;
  const float* ap = al + hh * D;
  const float* bp = ar + hh * D;
  float sl = 0.f, sr = 0.f;
  for (int d = 0; d < D; ++d) {
    float t = hp[d];
    sl = fmaf(t, ap[d], sl);
    sr = fmaf(t, bp[d], sr);
  }
  el[i] = sl;
  er[i] = sr;
}

// ---------------- edge pass 1: denominator of segment softmax ----------------
__global__ void gat_edge_esum(const int* __restrict__ ei, const float* __restrict__ el,
                              const float* __restrict__ er, float* __restrict__ esum,
                              int E, int H) {
  int e = blockIdx.x * blockDim.x + threadIdx.x;
  if (e >= E) return;
  int s = ei[e], d = ei[E + e];
  for (int h = 0; h < H; ++h) {
    float xv = el[s * H + h] + er[d * H + h];
    xv = xv > 0.f ? xv : 0.2f * xv;
    atomicAdd(&esum[d * H + h], __expf(xv));
  }
}

// ---------------- edge pass 2 (layers 1/2): scatter h[src]*alpha ----------------
__global__ __launch_bounds__(256) void gat_edge_msg256(const int* __restrict__ ei,
    const float* __restrict__ el, const float* __restrict__ er,
    const float* __restrict__ esum, const float* __restrict__ h,
    float* __restrict__ agg, int E) {
  int e = blockIdx.x;
  int t = threadIdx.x;
  int hh = t >> 6;
  int s = ei[e], d = ei[E + e];
  float xv = el[s * HEADS + hh] + er[d * HEADS + hh];
  xv = xv > 0.f ? xv : 0.2f * xv;
  float alpha = __expf(xv) / esum[d * HEADS + hh];
  atomicAdd(&agg[(size_t)d * HD + t], h[(size_t)s * HD + t] * alpha);
}

// ---------------- edge pass 2 (layer 3, H=1, D=40) ----------------
__global__ __launch_bounds__(64) void gat_edge_msg40(const int* __restrict__ ei,
    const float* __restrict__ el, const float* __restrict__ er,
    const float* __restrict__ esum, const float* __restrict__ h,
    float* __restrict__ agg, int E) {
  int e = blockIdx.x;
  int t = threadIdx.x;
  int s = ei[e], d = ei[E + e];
  float xv = el[s] + er[d];
  xv = xv > 0.f ? xv : 0.2f * xv;
  float alpha = __expf(xv) / esum[d];
  if (t < NCLASS)
    atomicAdd(&agg[(size_t)d * NCLASS + t], h[(size_t)s * NCLASS_PAD + t] * alpha);
}

// ---------------- bias + ELU + cast to bf16 (next layer input) ----------------
__global__ void gat_bias_elu_tobf16(const float* __restrict__ agg,
                                    const float* __restrict__ b,
                                    __bf16* __restrict__ featb, int total) {
  int i = blockIdx.x * blockDim.x + threadIdx.x;
  if (i >= total) return;
  float v = agg[i] + b[i & (HD - 1)];
  v = v > 0.f ? v : (__expf(v) - 1.f);
  featb[i] = (__bf16)v;
}

// ---------------- layer3 bias + log_softmax ----------------
__global__ void gat_final3(const float* __restrict__ agg3, const float* __restrict__ b3,
                           float* __restrict__ out, int N) {
  int n = blockIdx.x * blockDim.x + threadIdx.x;
  if (n >= N) return;
  float v[NCLASS];
  float mx = -3.4e38f;
  for (int c = 0; c < NCLASS; ++c) {
    float t = agg3[(size_t)n * NCLASS + c] + b3[c];
    v[c] = t;
    mx = fmaxf(mx, t);
  }
  float s = 0.f;
  for (int c = 0; c < NCLASS; ++c) s += __expf(v[c] - mx);
  float lse = mx + __logf(s);
  for (int c = 0; c < NCLASS; ++c) out[(size_t)n * NCLASS + c] = v[c] - lse;
}

extern "C" void kernel_launch(void* const* d_in, const int* in_sizes, int n_in,
                              void* d_out, int out_size, void* d_ws, size_t ws_size,
                              hipStream_t stream) {
  const float* x   = (const float*)d_in[0];
  const int*   ei  = (const int*)d_in[1];
  const float* W1  = (const float*)d_in[2];
  const float* al1 = (const float*)d_in[3];
  const float* ar1 = (const float*)d_in[4];
  const float* b1  = (const float*)d_in[5];
  const float* W2  = (const float*)d_in[6];
  const float* al2 = (const float*)d_in[7];
  const float* ar2 = (const float*)d_in[8];
  const float* b2  = (const float*)d_in[9];
  const float* W3  = (const float*)d_in[10];
  const float* al3 = (const float*)d_in[11];
  const float* ar3 = (const float*)d_in[12];
  const float* b3  = (const float*)d_in[13];
  float* out = (float*)d_out;

  // ---- carve workspace ----
  char* wsp = (char*)d_ws;
  auto carve = [&](size_t bytes) -> void* {
    void* p = (void*)wsp;
    wsp += (bytes + 511) & ~(size_t)511;
    return p;
  };
  const int N = N_NODES, E = N_EDGES;
  __bf16* featb = (__bf16*)carve((size_t)N * HD * 2);   // current layer input, bf16
  float*  h     = (float*)carve((size_t)N * HD * 4);    // GEMM output
  float*  agg   = (float*)carve((size_t)N * HD * 4);    // aggregation accumulator
  float*  el    = (float*)carve((size_t)N * HEADS * 4);
  float*  er    = (float*)carve((size_t)N * HEADS * 4);
  float*  esum  = (float*)carve((size_t)N * HEADS * 4);
  __bf16* Wp1   = (__bf16*)carve((size_t)256 * 256 * 2);
  __bf16* Wp2   = (__bf16*)carve((size_t)256 * 256 * 2);
  __bf16* Wp3   = (__bf16*)carve((size_t)256 * NCLASS_PAD * 2);
  float*  agg3  = agg;                                  // alias (8MB < 51MB)

  const int TPB = 256;
  int nhd_blocks = (N * HD + TPB - 1) / TPB;

  // ---- pack inputs/weights to bf16 / fragment layout ----
  gat_f32_to_bf16<<<nhd_blocks, TPB, 0, stream>>>(x, featb, N * HD);
  gat_pack_w<<<(256 * 256 + TPB - 1) / TPB, TPB, 0, stream>>>(W1, Wp1, 256, 256, 256);
  gat_pack_w<<<(256 * 256 + TPB - 1) / TPB, TPB, 0, stream>>>(W2, Wp2, 256, 256, 256);
  gat_pack_w<<<(256 * NCLASS_PAD + TPB - 1) / TPB, TPB, 0, stream>>>(W3, Wp3, NCLASS, NCLASS_PAD, 256);

  int tiles256 = (N / 16) * (256 / 64);
  int gemm256_blocks = (tiles256 + 7) / 8;
  int tiles64 = (N / 16) * (NCLASS_PAD / 64);
  int gemm64_blocks = (tiles64 + 7) / 8;
  int e_blocks = (E + TPB - 1) / TPB;
  int nh_blocks = (N * HEADS + TPB - 1) / TPB;
  int n_blocks = (N + TPB - 1) / TPB;

  // ================ Layer 1 ================
  gat_gemm_bf16<<<gemm256_blocks, TPB, 0, stream>>>(featb, Wp1, h, N, 256);
  gat_eler<<<nh_blocks, TPB, 0, stream>>>(h, al1, ar1, el, er, N, HEADS, HID, HD);
  hipMemsetAsync(esum, 0, (size_t)N * HEADS * 4, stream);
  gat_edge_esum<<<e_blocks, TPB, 0, stream>>>(ei, el, er, esum, E, HEADS);
  hipMemsetAsync(agg, 0, (size_t)N * HD * 4, stream);
  gat_edge_msg256<<<E, TPB, 0, stream>>>(ei, el, er, esum, h, agg, E);
  gat_bias_elu_tobf16<<<nhd_blocks, TPB, 0, stream>>>(agg, b1, featb, N * HD);

  // ================ Layer 2 ================
  gat_gemm_bf16<<<gemm256_blocks, TPB, 0, stream>>>(featb, Wp2, h, N, 256);
  gat_eler<<<nh_blocks, TPB, 0, stream>>>(h, al2, ar2, el, er, N, HEADS, HID, HD);
  hipMemsetAsync(esum, 0, (size_t)N * HEADS * 4, stream);
  gat_edge_esum<<<e_blocks, TPB, 0, stream>>>(ei, el, er, esum, E, HEADS);
  hipMemsetAsync(agg, 0, (size_t)N * HD * 4, stream);
  gat_edge_msg256<<<E, TPB, 0, stream>>>(ei, el, er, esum, h, agg, E);
  gat_bias_elu_tobf16<<<nhd_blocks, TPB, 0, stream>>>(agg, b2, featb, N * HD);

  // ================ Layer 3 (1 head, 40 classes padded to 64) ================
  gat_gemm_bf16<<<gemm64_blocks, TPB, 0, stream>>>(featb, Wp3, h, N, NCLASS_PAD);
  gat_eler<<<n_blocks, TPB, 0, stream>>>(h, al3, ar3, el, er, N, 1, NCLASS, NCLASS_PAD);
  hipMemsetAsync(esum, 0, (size_t)N * 4, stream);
  gat_edge_esum<<<e_blocks, TPB, 0, stream>>>(ei, el, er, esum, E, 1);
  hipMemsetAsync(agg3, 0, (size_t)N * NCLASS * 4, stream);
  gat_edge_msg40<<<E, 64, 0, stream>>>(ei, el, er, esum, h, agg3, E);
  gat_final3<<<n_blocks, TPB, 0, stream>>>(agg3, b3, out, N);
}
